// MOEBlock_66468913873692
// MI455X (gfx1250) — matrix operand order repaired and checked by reading, not compiled, not run on hardware
//
#include <hip/hip_runtime.h>
#include <math.h>

// Problem constants (from reference)
#define EXPERTS 8
#define TOPK 2
#define HDIM 1024
#define IDIM 4096
#define TTOK 4096   // B*S = 2*2048

typedef __attribute__((ext_vector_type(16))) _Float16 v16h;
typedef __attribute__((ext_vector_type(8)))  float    v8f;

union F16x16 { v16h v; uint4 q[2]; };
union H4 { _Float16 h[4]; uint2 u; };

#if __has_builtin(__builtin_amdgcn_rcpf)
#define RCPF(x) __builtin_amdgcn_rcpf(x)
#else
#define RCPF(x) (1.0f / (x))
#endif

// Load a 16-half WMMA fragment from two 16B-aligned LDS chunks.
__device__ __forceinline__ v16h ld_frag16(const _Float16* p0, const _Float16* p1) {
    F16x16 f;
    f.q[0] = *(const uint4*)p0;
    f.q[1] = *(const uint4*)p1;
    return f.v;
}

__device__ __forceinline__ void store4h(_Float16* dst, float4 v) {
    H4 t;
    t.h[0] = (_Float16)v.x; t.h[1] = (_Float16)v.y;
    t.h[2] = (_Float16)v.z; t.h[3] = (_Float16)v.w;
    *(uint2*)dst = t.u;
}

// Async copy 16B global -> LDS (gfx1250). Tracked by ASYNCcnt.
__device__ __forceinline__ void async_b128(uint32_t lds_off, const void* gaddr) {
    asm volatile("global_load_async_to_lds_b128 %0, %1, off"
                 :: "v"(lds_off), "v"(gaddr) : "memory");
}
__device__ __forceinline__ void wait_async0() {
    asm volatile("s_wait_asynccnt 0x0" ::: "memory");
}

// ---------------------------------------------------------------------------
// 1) Convert x (fp32) -> xh (f16), whole tensor, vectorized.
// ---------------------------------------------------------------------------
__global__ __launch_bounds__(256) void convert_x_kernel(const float* __restrict__ x,
                                                        _Float16* __restrict__ xh) {
    size_t idx = (size_t)blockIdx.x * blockDim.x + threadIdx.x;  // T*H/4 threads
    size_t base = idx * 4;
    float4 v = *(const float4*)(x + base);
    store4h(xh + base, v);
}

// ---------------------------------------------------------------------------
// 2) Router: one wave per token. logits = x @ router_w + b, softmax, top-2,
//    write comb[T,E] (zeros except the two selected experts).
// ---------------------------------------------------------------------------
__global__ __launch_bounds__(256) void router_kernel(const float* __restrict__ x,
                                                     const float* __restrict__ rw,
                                                     const float* __restrict__ rb,
                                                     float* __restrict__ comb) {
    int lane = threadIdx.x & 31;
    int t = blockIdx.x * 8 + (threadIdx.x >> 5);
    const float* xr = x + (size_t)t * HDIM;
    float acc[EXPERTS];
#pragma unroll
    for (int e = 0; e < EXPERTS; ++e) acc[e] = 0.0f;
    for (int j = 0; j < HDIM / 32; ++j) {
        int h = lane + j * 32;
        float xv = xr[h];
        const float* w = rw + (size_t)h * EXPERTS;
#pragma unroll
        for (int e = 0; e < EXPERTS; ++e) acc[e] = fmaf(xv, w[e], acc[e]);
    }
#pragma unroll
    for (int m = 16; m >= 1; m >>= 1) {
#pragma unroll
        for (int e = 0; e < EXPERTS; ++e) acc[e] += __shfl_xor(acc[e], m, 32);
    }
    if (lane == 0) {
        float p[EXPERTS];
        float mx = -1e30f;
#pragma unroll
        for (int e = 0; e < EXPERTS; ++e) { p[e] = acc[e] + rb[e]; mx = fmaxf(mx, p[e]); }
        float s = 0.0f;
#pragma unroll
        for (int e = 0; e < EXPERTS; ++e) { p[e] = __expf(p[e] - mx); s += p[e]; }
        float inv = RCPF(s);
#pragma unroll
        for (int e = 0; e < EXPERTS; ++e) p[e] *= inv;
        int i1 = 0;
#pragma unroll
        for (int e = 1; e < EXPERTS; ++e) if (p[e] > p[i1]) i1 = e;
        int i2 = (i1 == 0) ? 1 : 0;
#pragma unroll
        for (int e = 0; e < EXPERTS; ++e) if (e != i1 && p[e] > p[i2]) i2 = e;
#pragma unroll
        for (int e = 0; e < EXPERTS; ++e) {
            float c = (e == i1) ? p[i1] : ((e == i2) ? p[i2] : 0.0f);
            comb[(size_t)t * EXPERTS + e] = c;
        }
    }
}

// ---------------------------------------------------------------------------
// 3) Deterministic per-expert token compaction (token-ascending order).
// ---------------------------------------------------------------------------
__global__ __launch_bounds__(256) void compact_kernel(const float* __restrict__ comb,
                                                      int* __restrict__ tok,
                                                      int* __restrict__ cnt) {
    int e = blockIdx.x;
    int tid = threadIdx.x;
    const int PER = TTOK / 256;  // 16
    __shared__ int ssum[256];
    int base = tid * PER;
    int flags[PER];
    int c = 0;
#pragma unroll
    for (int i = 0; i < PER; ++i) {
        flags[i] = (comb[(size_t)(base + i) * EXPERTS + e] != 0.0f) ? 1 : 0;
        c += flags[i];
    }
    ssum[tid] = c;
    __syncthreads();
    for (int off = 1; off < 256; off <<= 1) {
        int v = 0;
        if (tid >= off) v = ssum[tid - off];
        __syncthreads();
        ssum[tid] += v;
        __syncthreads();
    }
    int excl = ssum[tid] - c;
    int total = ssum[255];
    int pos = excl;
#pragma unroll
    for (int i = 0; i < PER; ++i) {
        if (flags[i]) tok[(size_t)e * TTOK + (pos++)] = base + i;
    }
    if (tid == 0) cnt[e] = total;
    for (int p = total + tid; p < TTOK; p += 256) tok[(size_t)e * TTOK + p] = 0;
}

// ---------------------------------------------------------------------------
// 4) Fused gate+up GEMM + SiLU(g*u), LDS double-buffered, async A-tiles.
//    Block tile 128x128; 8 waves, wave (wm,wn) owns 64x32 -> 4x2 WMMA tiles x2.
// ---------------------------------------------------------------------------
__global__ __launch_bounds__(256) void gateup_kernel(const float* __restrict__ gate_w,
                                                     const float* __restrict__ gate_b,
                                                     const float* __restrict__ up_w,
                                                     const float* __restrict__ up_b,
                                                     const _Float16* __restrict__ xh,
                                                     const int* __restrict__ tok,
                                                     const int* __restrict__ cnt,
                                                     _Float16* __restrict__ act,
                                                     int e) {
    const int cnt_e = cnt[e];
    const int bm = blockIdx.x;
    if (bm * 128 >= cnt_e) return;          // block-uniform: EXEC stays full
    const int n0 = blockIdx.y * 128;

    const int tid = threadIdx.x;
    const int lane = tid & 31;
    const int wave = tid >> 5;
    const int wm = wave & 1;
    const int wn = wave >> 1;

    __shared__ __align__(16) _Float16 As[2][128 * 40];
    __shared__ __align__(16) _Float16 Bsg[2][32 * 136];
    __shared__ __align__(16) _Float16 Bsu[2][32 * 136];

    const int rA = tid >> 1;
    const int part = tid & 1;
    const int tokenA = tok[(size_t)e * TTOK + bm * 128 + rA];
    const _Float16* xrow = xh + (size_t)tokenA * HDIM + part * 16;
    const float* gwp = gate_w + (size_t)e * HDIM * IDIM + n0;
    const float* uwp = up_w + (size_t)e * HDIM * IDIM + n0;

    auto issueA = [&](int k0, int b) {
        uint32_t lds = (uint32_t)(uintptr_t)&As[b][rA * 40 + part * 16];
        const _Float16* g0 = xrow + k0;
        async_b128(lds, g0);
        async_b128(lds + 16, g0 + 8);
    };
    float4 vg[4], vu[4];
    auto loadB = [&](int k0) {
#pragma unroll
        for (int j = 0; j < 4; ++j) {
            int f = tid + j * 256;
            int row = f >> 5;
            int c4 = f & 31;
            vg[j] = *(const float4*)(gwp + (size_t)(k0 + row) * IDIM + c4 * 4);
            vu[j] = *(const float4*)(uwp + (size_t)(k0 + row) * IDIM + c4 * 4);
        }
    };
    auto storeB = [&](int b) {
#pragma unroll
        for (int j = 0; j < 4; ++j) {
            int f = tid + j * 256;
            int row = f >> 5;
            int c4 = f & 31;
            store4h(&Bsg[b][row * 136 + c4 * 4], vg[j]);
            store4h(&Bsu[b][row * 136 + c4 * 4], vu[j]);
        }
    };

    v8f accg[4][2];
    v8f accu[4][2];
#pragma unroll
    for (int mi = 0; mi < 4; ++mi)
#pragma unroll
        for (int ni = 0; ni < 2; ++ni) { accg[mi][ni] = (v8f)(0.0f); accu[mi][ni] = (v8f)(0.0f); }

    // prologue: stage tile 0
    issueA(0, 0);
    loadB(0);
    storeB(0);
    wait_async0();
    __syncthreads();

    int buf = 0;
    for (int k0 = 0; k0 < HDIM; k0 += 32) {
        const bool has_next = (k0 + 32 < HDIM);
        if (has_next) {
            issueA(k0 + 32, buf ^ 1);
            loadB(k0 + 32);
        }
        v16h a[4];
#pragma unroll
        for (int mi = 0; mi < 4; ++mi) {
            const _Float16* pa = &As[buf][(wm * 64 + mi * 16 + (lane & 15)) * 40 + ((lane >> 4) * 8)];
            a[mi] = ld_frag16(pa, pa + 16);
        }
#pragma unroll
        for (int ni = 0; ni < 2; ++ni) {
            const _Float16* pbg = &Bsg[buf][wn * 32 + ni * 16 + (lane & 15)];
            const _Float16* pbu = &Bsu[buf][wn * 32 + ni * 16 + (lane & 15)];
            v16h bg, bu;
#pragma unroll
            for (int s = 0; s < 16; ++s) {
                int kk = ((s >> 3) << 4) + ((lane >> 4) << 3) + (s & 7);
                bg[s] = pbg[kk * 136];
                bu[s] = pbu[kk * 136];
            }
#pragma unroll
            for (int mi = 0; mi < 4; ++mi) {
                accg[mi][ni] = __builtin_amdgcn_wmma_f32_16x16x32_f16(
                    false, a[mi], false, bg, (short)0, accg[mi][ni], false, false);
                accu[mi][ni] = __builtin_amdgcn_wmma_f32_16x16x32_f16(
                    false, a[mi], false, bu, (short)0, accu[mi][ni], false, false);
            }
        }
        if (has_next) {
            storeB(buf ^ 1);
            wait_async0();
        }
        __syncthreads();
        buf ^= 1;
    }

    // epilogue: act = silu((g+gb)*(u+ub)) in f16
    const int rowbase = bm * 128 + wm * 64;
    const int colbase = n0 + wn * 32;
    const int rowsel = (lane >> 4) * 8;
    const int collane = lane & 15;
    const bool full = (bm * 128 + 128 <= cnt_e);
#pragma unroll
    for (int ni = 0; ni < 2; ++ni) {
        int cg = colbase + ni * 16 + collane;
        float gb = gate_b[(size_t)e * IDIM + cg];
        float ub = up_b[(size_t)e * IDIM + cg];
#pragma unroll
        for (int mi = 0; mi < 4; ++mi) {
#pragma unroll
            for (int v = 0; v < 8; ++v) {
                int r = rowbase + mi * 16 + rowsel + v;
                if (full || r < cnt_e) {
                    float g = accg[mi][ni][v] + gb;
                    float u = accu[mi][ni][v] + ub;
                    float z = g * u;
                    float sv = z * RCPF(1.0f + __expf(-z));
                    act[(size_t)r * IDIM + cg] = (_Float16)sv;
                }
            }
        }
    }
}

// ---------------------------------------------------------------------------
// 5) Down GEMM + weighted scatter-add, LDS double-buffered, async A-tiles.
// ---------------------------------------------------------------------------
__global__ __launch_bounds__(256) void down_kernel(const float* __restrict__ down_w,
                                                   const float* __restrict__ down_b,
                                                   const _Float16* __restrict__ act,
                                                   const int* __restrict__ tok,
                                                   const int* __restrict__ cnt,
                                                   const float* __restrict__ comb,
                                                   float* __restrict__ out,
                                                   int e) {
    const int cnt_e = cnt[e];
    const int bm = blockIdx.x;
    if (bm * 128 >= cnt_e) return;
    const int n0 = blockIdx.y * 128;

    const int tid = threadIdx.x;
    const int lane = tid & 31;
    const int wave = tid >> 5;
    const int wm = wave & 1;
    const int wn = wave >> 1;

    __shared__ __align__(16) _Float16 As[2][128 * 40];
    __shared__ __align__(16) _Float16 Bs[2][32 * 136];

    const int rA = tid >> 1;
    const int part = tid & 1;
    const _Float16* arow = act + (size_t)(bm * 128 + rA) * IDIM + part * 16;
    const float* dwp = down_w + (size_t)e * IDIM * HDIM + n0;

    auto issueA = [&](int k0, int b) {
        uint32_t lds = (uint32_t)(uintptr_t)&As[b][rA * 40 + part * 16];
        const _Float16* g0 = arow + k0;
        async_b128(lds, g0);
        async_b128(lds + 16, g0 + 8);
    };
    float4 vd[4];
    auto loadB = [&](int k0) {
#pragma unroll
        for (int j = 0; j < 4; ++j) {
            int f = tid + j * 256;
            int row = f >> 5;
            int c4 = f & 31;
            vd[j] = *(const float4*)(dwp + (size_t)(k0 + row) * HDIM + c4 * 4);
        }
    };
    auto storeB = [&](int b) {
#pragma unroll
        for (int j = 0; j < 4; ++j) {
            int f = tid + j * 256;
            int row = f >> 5;
            int c4 = f & 31;
            store4h(&Bs[b][row * 136 + c4 * 4], vd[j]);
        }
    };

    v8f acc[4][2];
#pragma unroll
    for (int mi = 0; mi < 4; ++mi)
#pragma unroll
        for (int ni = 0; ni < 2; ++ni) acc[mi][ni] = (v8f)(0.0f);

    issueA(0, 0);
    loadB(0);
    storeB(0);
    wait_async0();
    __syncthreads();

    int buf = 0;
    for (int k0 = 0; k0 < IDIM; k0 += 32) {
        const bool has_next = (k0 + 32 < IDIM);
        if (has_next) {
            issueA(k0 + 32, buf ^ 1);
            loadB(k0 + 32);
        }
        v16h a[4];
#pragma unroll
        for (int mi = 0; mi < 4; ++mi) {
            const _Float16* pa = &As[buf][(wm * 64 + mi * 16 + (lane & 15)) * 40 + ((lane >> 4) * 8)];
            a[mi] = ld_frag16(pa, pa + 16);
        }
#pragma unroll
        for (int ni = 0; ni < 2; ++ni) {
            const _Float16* pb = &Bs[buf][wn * 32 + ni * 16 + (lane & 15)];
            v16h b;
#pragma unroll
            for (int s = 0; s < 16; ++s) {
                int kk = ((s >> 3) << 4) + ((lane >> 4) << 3) + (s & 7);
                b[s] = pb[kk * 136];
            }
#pragma unroll
            for (int mi = 0; mi < 4; ++mi) {
                acc[mi][ni] = __builtin_amdgcn_wmma_f32_16x16x32_f16(
                    false, a[mi], false, b, (short)0, acc[mi][ni], false, false);
            }
        }
        if (has_next) {
            storeB(buf ^ 1);
            wait_async0();
        }
        __syncthreads();
        buf ^= 1;
    }

    const int rowbase = bm * 128 + wm * 64;
    const int colbase = n0 + wn * 32;
    const int rowsel = (lane >> 4) * 8;
    const int collane = lane & 15;
    const bool full = (bm * 128 + 128 <= cnt_e);
#pragma unroll
    for (int ni = 0; ni < 2; ++ni) {
        int cg = colbase + ni * 16 + collane;
        float db = down_b[(size_t)e * HDIM + cg];
#pragma unroll
        for (int mi = 0; mi < 4; ++mi) {
#pragma unroll
            for (int v = 0; v < 8; ++v) {
                int r = rowbase + mi * 16 + rowsel + v;
                if (full || r < cnt_e) {
                    int tk = tok[(size_t)e * TTOK + r];
                    float w = comb[(size_t)tk * EXPERTS + e];
                    float* o = out + (size_t)tk * HDIM + cg;
                    *o += (acc[mi][ni][v] + db) * w;
                }
            }
        }
    }
}

// ---------------------------------------------------------------------------
// Host side
// ---------------------------------------------------------------------------
extern "C" void kernel_launch(void* const* d_in, const int* in_sizes, int n_in,
                              void* d_out, int out_size, void* d_ws, size_t ws_size,
                              hipStream_t stream) {
    const float* x  = (const float*)d_in[0];
    const float* rw = (const float*)d_in[1];
    const float* rb = (const float*)d_in[2];
    const float* gw = (const float*)d_in[3];
    const float* gb = (const float*)d_in[4];
    const float* uw = (const float*)d_in[5];
    const float* ub = (const float*)d_in[6];
    const float* dw = (const float*)d_in[7];
    const float* db = (const float*)d_in[8];
    float* out = (float*)d_out;

    char* ws = (char*)d_ws;
    float*    comb = (float*)ws;                          // 131072 B
    int*      tok  = (int*)(ws + 131072);                 // 131072 B
    int*      cnt  = (int*)(ws + 262144);                 // 128 B
    _Float16* xh   = (_Float16*)(ws + 262272);            // 8 MiB
    _Float16* act  = (_Float16*)(ws + 262272 + 8388608);  // 32 MiB

    hipMemsetAsync(d_out, 0, (size_t)out_size * sizeof(float), stream);

    convert_x_kernel<<<(TTOK * HDIM) / (256 * 4), 256, 0, stream>>>(x, xh);
    router_kernel<<<TTOK / 8, 256, 0, stream>>>(x, rw, rb, comb);
    compact_kernel<<<EXPERTS, 256, 0, stream>>>(comb, tok, cnt);

    for (int e = 0; e < EXPERTS; ++e) {
        gateup_kernel<<<dim3(TTOK / 128, IDIM / 128), 256, 0, stream>>>(
            gw, gb, uw, ub, xh, tok, cnt, act, e);
        down_kernel<<<dim3(TTOK / 128, HDIM / 128), 256, 0, stream>>>(
            dw, db, act, tok, cnt, comb, out, e);
    }
}
